// NonLocal_54030688584206
// MI455X (gfx1250) — compile-verified, matrix operand order checked by reading, not run on hardware
//
#include <hip/hip_runtime.h>
#include <hip/hip_bf16.h>

#define NPOS 4096   // D*H*W = 16*16*16
#define CIN  128    // input channels
#define CHP  64     // projected channels
#define NB   2      // batch

typedef __attribute__((ext_vector_type(16))) __bf16 v16bf;
typedef __attribute__((ext_vector_type(8)))  float  v8f;

union Frag16 {
    v16bf v;
    uint4 q[2];          // 2 x 16 bytes = 16 bf16
};

__device__ __forceinline__ unsigned short f2bf(float f) {
    union { float f; unsigned u; } c; c.f = f;
    unsigned r = c.u + 0x7FFFu + ((c.u >> 16) & 1u);   // round-to-nearest-even
    return (unsigned short)(r >> 16);
}

// Async copy 16 bytes per lane: global -> LDS, tracked by ASYNCcnt.
// ISA 10.x / 15.18: GLOBAL_LOAD_ASYNC_TO_LDS_B128, GV addressing mode.
__device__ __forceinline__ void async_b128(unsigned lds_off, const void* g) {
    asm volatile("global_load_async_to_lds_b128 %0, %1, off"
                 :: "v"(lds_off), "v"(g)
                 : "memory");
}
__device__ __forceinline__ void wait_async0() {
    asm volatile("s_wait_asynccnt 0x0" ::: "memory");
}

// Load a 16x32 bf16 operand fragment (A or B style) from a row-major matrix
// (works for global or LDS-backed pointers; inlined so AS is inferred).
// 16-bit operand layout (ISA 7.12.2): lane l -> row = row0 + (l&15);
// lanes 0-15 hold K = {0..7, 16..23}, lanes 16-31 hold K = {8..15, 24..31}.
__device__ __forceinline__ v16bf load_frag_rm(const unsigned short* base,
                                              int row0, int col0, int ld) {
    int l = threadIdx.x & 31;
    const unsigned short* p = base + (size_t)(row0 + (l & 15)) * ld
                                   + col0 + ((l >> 4) << 3);
    Frag16 f;
    f.q[0] = *(const uint4*)(p);
    f.q[1] = *(const uint4*)(p + 16);
    return f.v;
}

__device__ __forceinline__ v8f wmma_bf16(v16bf a, v16bf b, v8f c) {
    return __builtin_amdgcn_wmma_f32_16x16x32_bf16(false, a, false, b,
                                                   (short)0, c, false, false);
}

// ---------------------------------------------------------------------------
// K1: q/k/v projections (1x1x1 conv == channel matmul), fp32 -> bf16.
// qT,kT stored (NPOS x CHP) row-major, v stored (CHP x NPOS) row-major.
// ---------------------------------------------------------------------------
__global__ __launch_bounds__(256)
void proj_kernel(const float* __restrict__ x,
                 const float* __restrict__ Wq,
                 const float* __restrict__ Wk,
                 const float* __restrict__ Wv,
                 unsigned short* __restrict__ qT,
                 unsigned short* __restrict__ kT,
                 unsigned short* __restrict__ vb) {
    int p = blockIdx.x * 256 + threadIdx.x;
    int o = blockIdx.y;
    int n = blockIdx.z;
    const float* xb = x + (size_t)n * CIN * NPOS + p;
    float aq = 0.f, ak = 0.f, av = 0.f;
    #pragma unroll 8
    for (int c = 0; c < CIN; ++c) {
        float xv = xb[(size_t)c * NPOS];
        aq = fmaf(Wq[o * CIN + c], xv, aq);
        ak = fmaf(Wk[o * CIN + c], xv, ak);
        av = fmaf(Wv[o * CIN + c], xv, av);
    }
    qT[((size_t)n * NPOS + p) * CHP + o] = f2bf(aq);
    kT[((size_t)n * NPOS + p) * CHP + o] = f2bf(ak);
    vb[((size_t)n * CHP + o) * NPOS + p] = f2bf(av);
}

// ---------------------------------------------------------------------------
// K2: row statistics  m[i] = max_j s[i,j],  Z[i] = sum_j exp(s[i,j]-m[i]).
// Block = 8 waves, each owns 16 rows i.  The 32xCHP q tile of each j-step is
// async-staged into LDS once per block (all waves share it), then each wave
// runs 2x(2 chained bf16 WMMAs) and an online max/sum update.  Final 4-step
// shfl_xor merge within each 16-lane half; writes (m, 1/Z) as float2.
// ---------------------------------------------------------------------------
__global__ __launch_bounds__(256)
void stats_kernel(const unsigned short* __restrict__ qT,
                  const unsigned short* __restrict__ kT,
                  float2* __restrict__ stats) {
    __shared__ __align__(16) unsigned short qbuf[32 * CHP];   // 4 KB shared q tile
    int tid  = threadIdx.x;
    int wave = tid >> 5;
    int lane = tid & 31;
    int n    = blockIdx.y;
    int i0   = blockIdx.x * 128 + wave * 16;

    const unsigned short* kTb = kT + (size_t)n * NPOS * CHP;
    const unsigned short* qTb = qT + (size_t)n * NPOS * CHP;

    v16bf a0 = load_frag_rm(kTb, i0, 0,  CHP);   // A: k^T rows i0..i0+15
    v16bf a1 = load_frag_rm(kTb, i0, 32, CHP);

    float m[8], s[8];
    #pragma unroll
    for (int r = 0; r < 8; ++r) { m[r] = -3.0e38f; s[r] = 0.f; }

    int crow = tid >> 3;            // 0..31 (j row in tile)
    int ccol = (tid & 7) * 8;       // halfword column
    unsigned qdst = (unsigned)(uintptr_t)&qbuf[crow * CHP + ccol];

    for (int j0 = 0; j0 < NPOS; j0 += 32) {
        async_b128(qdst, qTb + (size_t)(j0 + crow) * CHP + ccol);
        wait_async0();
        __syncthreads();
        #pragma unroll
        for (int jt = 0; jt < 2; ++jt) {
            v16bf b0 = load_frag_rm(qbuf, jt * 16, 0,  CHP);
            v16bf b1 = load_frag_rm(qbuf, jt * 16, 32, CHP);
            v8f d = {};
            d = wmma_bf16(a0, b0, d);
            d = wmma_bf16(a1, b1, d);
            #pragma unroll
            for (int r = 0; r < 8; ++r) {
                float v  = d[r];
                float nm = fmaxf(m[r], v);
                s[r] = s[r] * __expf(m[r] - nm) + __expf(v - nm);
                m[r] = nm;
            }
        }
        __syncthreads();
    }
    // merge across the 16 lanes (= j columns) inside each half-wave
    #pragma unroll
    for (int off = 1; off < 16; off <<= 1) {
        #pragma unroll
        for (int r = 0; r < 8; ++r) {
            float om = __shfl_xor(m[r], off, 32);
            float os = __shfl_xor(s[r], off, 32);
            float nm = fmaxf(m[r], om);
            s[r] = s[r] * __expf(m[r] - nm) + os * __expf(om - nm);
            m[r] = nm;
        }
    }
    if ((lane & 15) == 0) {
        int rbase = i0 + ((lane >> 4) << 3);
        #pragma unroll
        for (int r = 0; r < 8; ++r)
            stats[(size_t)n * NPOS + rbase + r] = make_float2(m[r], 1.f / s[r]);
    }
}

// ---------------------------------------------------------------------------
// K3: output pass.  Block = 8 waves, each owns 16 output columns j.  For each
// 32-row i-chunk: the block async-stages k^T (4KB), V (4KB) and stats (256B)
// into LDS once (all waves share them), prefetches the next chunk, then each
// wave recomputes its score tiles (WMMA), p = exp(s-m[i]) * (1/Z[i]), scatters
// p through a per-wave LDS region directly into the B-operand register image,
// and accumulates the 4 channel tiles:  acc(c) += V(16x32) x P(32x16).
// ---------------------------------------------------------------------------
__global__ __launch_bounds__(256)
void attn_kernel(const unsigned short* __restrict__ qT,
                 const unsigned short* __restrict__ kT,
                 const unsigned short* __restrict__ vb,
                 const float2* __restrict__ stats,
                 float* __restrict__ agg) {
    __shared__ __align__(16) unsigned short kbuf[32 * CHP];     // 4 KB k^T chunk
    __shared__ __align__(16) unsigned short vbuf[CHP * 32];     // 4 KB V chunk
    __shared__ __align__(16) float2         sbuf[32];           // 256 B stats
    __shared__ __align__(16) unsigned short pbuf[8 * 32 * 16];  // 1 KB / wave

    int tid  = threadIdx.x;
    int wave = tid >> 5;
    int lane = tid & 31;
    int n    = blockIdx.y;
    int j0   = blockIdx.x * 128 + wave * 16;

    const unsigned short* kTb = kT + (size_t)n * NPOS * CHP;
    const unsigned short* qTb = qT + (size_t)n * NPOS * CHP;
    const unsigned short* vbb = vb + (size_t)n * CHP * NPOS;
    const float2*         st  = stats + (size_t)n * NPOS;

    v16bf bq0 = load_frag_rm(qTb, j0, 0,  CHP);   // loop-invariant B of scores
    v16bf bq1 = load_frag_rm(qTb, j0, 32, CHP);

    v8f acc[4];
    #pragma unroll
    for (int t = 0; t < 4; ++t) acc[t] = (v8f){};

    unsigned short* L = pbuf + wave * (32 * 16);
    int col = lane & 15;
    int hi8 = (lane >> 4) << 3;

    // per-thread staging coordinates (constant over the loop)
    int krow = tid >> 3, kcol = (tid & 7) * 8;      // 32 x 64 tile
    int vrow = tid >> 2, vcol = (tid & 3) * 8;      // 64 x 32 tile
    unsigned kdst = (unsigned)(uintptr_t)&kbuf[krow * CHP + kcol];
    unsigned vdst = (unsigned)(uintptr_t)&vbuf[vrow * 32 + vcol];
    unsigned sdst = (unsigned)(uintptr_t)&sbuf[(tid & 15) * 2];

    for (int i0 = 0; i0 < NPOS; i0 += 32) {
        // cooperative async staging of shared operands for this i-chunk
        async_b128(kdst, kTb + (size_t)(i0 + krow) * CHP + kcol);
        async_b128(vdst, vbb + (size_t)vrow * NPOS + i0 + vcol);
        if (tid < 16) async_b128(sdst, (const void*)(st + i0 + tid * 2));
        wait_async0();
        __syncthreads();
        if (i0 + 32 < NPOS)   // warm L2 for the next chunk during compute
            __builtin_prefetch(kTb + (size_t)(i0 + 32 + krow) * CHP + kcol, 0, 3);

        #pragma unroll
        for (int t = 0; t < 2; ++t) {
            v16bf ak0 = load_frag_rm(kbuf, t * 16, 0,  CHP);
            v16bf ak1 = load_frag_rm(kbuf, t * 16, 32, CHP);
            v8f d = {};
            d = wmma_bf16(ak0, bq0, d);
            d = wmma_bf16(ak1, bq1, d);
            #pragma unroll
            for (int r = 0; r < 8; ++r) {
                int iloc = t * 16 + r + hi8;          // 0..31 within chunk
                float2 ms = sbuf[iloc];
                unsigned short pb = f2bf(__expf(d[r] - ms.x) * ms.y);
                // place at the B-fragment register-image slot of its consumer:
                int rl = col + (((iloc & 15) < 8) ? 0 : 16);      // reader lane
                int e  = (iloc & 7) + ((iloc >= 16) ? 8 : 0);     // element idx
                L[rl * 16 + e] = pb;
            }
        }
        asm volatile("s_wait_dscnt 0x0" ::: "memory");  // per-wave DS stores done
        Frag16 pf;
        pf.q[0] = *(const uint4*)(L + lane * 16);
        pf.q[1] = *(const uint4*)(L + lane * 16 + 8);
        v16bf pb = pf.v;
        #pragma unroll
        for (int t = 0; t < 4; ++t) {
            v16bf av = load_frag_rm(vbuf, t * 16, 0, 32);   // V 16c x 32i from LDS
            acc[t] = wmma_bf16(av, pb, acc[t]);
        }
        __syncthreads();   // shared kbuf/vbuf/sbuf reuse next chunk
    }

    #pragma unroll
    for (int t = 0; t < 4; ++t)
        #pragma unroll
        for (int r = 0; r < 8; ++r)
            agg[((size_t)n * CHP + t * 16 + r + hi8) * NPOS + j0 + col] = acc[t][r];
}

// ---------------------------------------------------------------------------
// K4: post projection (128x64) + residual add.
// ---------------------------------------------------------------------------
__global__ __launch_bounds__(256)
void post_kernel(const float* __restrict__ agg,
                 const float* __restrict__ Wpost,
                 const float* __restrict__ x,
                 float* __restrict__ out) {
    int p = blockIdx.x * 256 + threadIdx.x;
    int o = blockIdx.y;
    int n = blockIdx.z;
    const float* ab = agg + (size_t)n * CHP * NPOS + p;
    float a = 0.f;
    #pragma unroll 8
    for (int c = 0; c < CHP; ++c)
        a = fmaf(Wpost[o * CHP + c], ab[(size_t)c * NPOS], a);
    size_t idx = ((size_t)n * CIN + o) * NPOS + p;
    out[idx] = a + x[idx];
}

extern "C" void kernel_launch(void* const* d_in, const int* in_sizes, int n_in,
                              void* d_out, int out_size, void* d_ws, size_t ws_size,
                              hipStream_t stream) {
    const float* x     = (const float*)d_in[0];
    const float* Wq    = (const float*)d_in[1];
    const float* Wk    = (const float*)d_in[2];
    const float* Wv    = (const float*)d_in[3];
    const float* Wpost = (const float*)d_in[4];
    float* out = (float*)d_out;

    // workspace layout: qT 1MB | kT 1MB | v 1MB | stats 64KB | agg 2MB
    size_t nqkv = (size_t)NB * NPOS * CHP;
    unsigned short* qT = (unsigned short*)d_ws;
    unsigned short* kT = qT + nqkv;
    unsigned short* vb = kT + nqkv;
    float2* stats = (float2*)(vb + nqkv);
    float*  agg   = (float*)(stats + (size_t)NB * NPOS);

    proj_kernel <<<dim3(NPOS / 256, CHP, NB), 256, 0, stream>>>(x, Wq, Wk, Wv, qT, kT, vb);
    stats_kernel<<<dim3(NPOS / 128, NB),      256, 0, stream>>>(qT, kT, stats);
    attn_kernel <<<dim3(NPOS / 128, NB),      256, 0, stream>>>(qT, kT, vb, stats, agg);
    post_kernel <<<dim3(NPOS / 256, CIN, NB), 256, 0, stream>>>(agg, Wpost, x, out);
}
